// DeepGCN_31602369364483
// MI455X (gfx1250) — compile-verified, matrix operand order checked by reading, not run on hardware
//
#include <hip/hip_runtime.h>
#include <hip/hip_bf16.h>
#include <math.h>

// ---------------------------------------------------------------------------
// CDNA5 (gfx1250) DeepGCN pipeline, round 3.
//  - Activations stored as bf16: WMMA A-fragments are two global_load_b128
//    per K-step, row index clamped (not branch-guarded) so the inner loop has
//    no EXEC manipulation or register zero-fill.
//  - Dense GEMMs: wave32 v_wmma_f32_16x16x32_bf16; W staged once to LDS as
//    transposed bf16 so B-fragments are two ds_load_b128.
//  - GCN aggregation: f32 xw buffer, self-loop term initializes accumulator,
//    edges via hardware global_atomic_add_f32 (bandwidth-roofline path).
//  - bias+tanh fused into bf16-writing epilogues.
// ---------------------------------------------------------------------------

typedef __attribute__((ext_vector_type(16))) __bf16 v16bf;
typedef __attribute__((ext_vector_type(8)))  float  v8f;

union BF16Frag {
    v16bf v;
    unsigned short u[16];
    uint4 q[2];
};

__device__ __forceinline__ unsigned short f32_bf16(float f) {
    unsigned int u = __float_as_uint(f);
    u += 0x7FFFu + ((u >> 16) & 1u);          // round-to-nearest-even
    return (unsigned short)(u >> 16);
}
__device__ __forceinline__ float bf_lo(unsigned int w) { return __uint_as_float(w << 16); }
__device__ __forceinline__ float bf_hi(unsigned int w) { return __uint_as_float(w & 0xffff0000u); }

// ---------------------------------------------------------------------------
// WMMA GEMM:  C[M x Foutp] = A[M x lda] * W[Kact x Fout],  A in bf16.
// Block = 256 threads = 8 waves; each wave owns a 16-row tile, loops over
// 16-wide column tiles. W staged to LDS as bf16 transposed [Foutp][Kp].
// OOB rows are clamped on load (results discarded at the guarded writeback).
// fuse=0: write f32 to Cf. fuse=1: write bf16 tanh(x+bias) to Ch (0 on pads).
// ---------------------------------------------------------------------------
__global__ __launch_bounds__(256) void k_gemm_wmma(
    const unsigned short* __restrict__ A, int lda, int M,
    const float* __restrict__ W, int Kact, int Kp,
    int Fout, int Foutp,
    float* __restrict__ Cf, unsigned short* __restrict__ Ch, int ldc,
    const float* __restrict__ bias, int fuse)
{
    __shared__ unsigned short sW[8192];   // 16 KB: [Foutp][Kp] bf16, transposed
    const int tid = threadIdx.x;
    const int total = Foutp * Kp;
    for (int idx = tid; idx < total; idx += 256) {
        int n = idx / Kp;
        int k = idx - n * Kp;
        float w = (n < Fout && k < Kact) ? W[k * Fout + n] : 0.0f;
        sW[idx] = f32_bf16(w);
    }
    __syncthreads();

    const int wave = tid >> 5;            // wave32
    const int lane = tid & 31;
    const int half = lane >> 4;           // K-run selector per ISA A layout
    const int l16  = lane & 15;
    const int rowBase = blockIdx.x * 128 + wave * 16;
    const int r = rowBase + l16;
    const int rc = (r < M) ? r : (M - 1); // clamp: keeps loads in-bounds,
                                          // OOB rows never written back
    const unsigned short* arow = A + (size_t)rc * lda;

    const int nTiles = Foutp >> 4;
    for (int ct = 0; ct < nTiles; ++ct) {
        const int colBase = ct << 4;
        const int nn = colBase + l16;
        const unsigned short* wrow = &sW[nn * Kp];
        v8f acc = {0.f, 0.f, 0.f, 0.f, 0.f, 0.f, 0.f, 0.f};

        for (int k0 = 0; k0 < Kp; k0 += 32) {
            const int ka = k0 + half * 8;     // K runs: {ka..ka+7, ka+16..ka+23}
            const int kb = ka + 16;
            BF16Frag a, b;
            a.q[0] = *(const uint4*)(arow + ka);   // global_load_b128 (8 bf16)
            a.q[1] = *(const uint4*)(arow + kb);
            b.q[0] = *(const uint4*)(wrow + ka);   // ds_load_b128
            b.q[1] = *(const uint4*)(wrow + kb);

            acc = __builtin_amdgcn_wmma_f32_16x16x32_bf16(
                false, a.v, false, b.v, (short)0, acc, false, false);
        }

        // C layout: VGPR i -> row (rowBase + i + 8*half), col nn
        #pragma unroll
        for (int i = 0; i < 8; ++i) {
            int rr = rowBase + i + half * 8;
            if (rr < M) {
                float val = acc[i];
                if (fuse) {
                    unsigned short o = (nn < Fout)
                        ? f32_bf16(tanhf(val + bias[nn])) : (unsigned short)0;
                    Ch[(size_t)rr * ldc + nn] = o;
                } else {
                    Cf[(size_t)rr * ldc + nn] = val;
                }
            }
        }
    }
}

// --------------------------- f32 -> bf16 pass ------------------------------
__global__ __launch_bounds__(256) void k_cvt_bf16(
    const float* __restrict__ src, unsigned short* __restrict__ dst, int total4)
{
    int gid = blockIdx.x * 256 + threadIdx.x;
    if (gid >= total4) return;
    float4 v = ((const float4*)src)[gid];
    ushort4 o = { f32_bf16(v.x), f32_bf16(v.y), f32_bf16(v.z), f32_bf16(v.w) };
    ((ushort4*)dst)[gid] = o;
}

// --------------------------- degree / norm ---------------------------------
__global__ __launch_bounds__(256) void k_fill1(float* __restrict__ p, int n) {
    int i = blockIdx.x * 256 + threadIdx.x;
    if (i < n) p[i] = 1.0f;                 // self-loop contributes 1 to degree
}

__global__ __launch_bounds__(256) void k_degacc(const int* __restrict__ dst,
                                                float* __restrict__ deg, int E_) {
    int e = blockIdx.x * 256 + threadIdx.x;
    if (e < E_) atomicAdd(&deg[dst[e]], 1.0f);
}

__global__ __launch_bounds__(256) void k_dinv(const float* __restrict__ deg,
                                              float* __restrict__ dinv, int n) {
    int i = blockIdx.x * 256 + threadIdx.x;
    if (i < n) {
        float d = deg[i];
        dinv[i] = (d > 0.0f) ? rsqrtf(fmaxf(d, 1.0f)) : 0.0f;
    }
}

// ------------------------ GCN aggregation ----------------------------------
// agg[i] = dinv[i]^2 * xw[i]   (self-loop term, initializes accumulator)
__global__ __launch_bounds__(256) void k_selfinit(
    const float* __restrict__ xw, const float* __restrict__ dinv,
    float* __restrict__ agg, int total4, int lg4)
{
    int gid = blockIdx.x * 256 + threadIdx.x;
    if (gid >= total4) return;
    int i = gid >> lg4;
    float w = dinv[i] * dinv[i];
    float4 v = ((const float4*)xw)[gid];
    float4 o = { w * v.x, w * v.y, w * v.z, w * v.w };
    ((float4*)agg)[gid] = o;
}

// agg[dst] += dinv[src]*dinv[dst] * xw[src]   (one thread = one edge x 4 feats)
__global__ __launch_bounds__(256) void k_scatter(
    const float* __restrict__ xw, const int* __restrict__ ei,
    const float* __restrict__ dinv, float* __restrict__ agg,
    int E_, int Fp4, int lg4, int total)
{
    int gid = blockIdx.x * 256 + threadIdx.x;
    if (gid >= total) return;
    int e = gid >> lg4;
    int c = gid & (Fp4 - 1);
    int s = ei[e];
    int d = ei[E_ + e];
    float w = dinv[s] * dinv[d];
    float4 v = ((const float4*)xw)[((size_t)s << lg4) + c];
    float* ap = agg + ((((size_t)d << lg4) + c) << 2);
    atomicAdd(ap + 0, w * v.x);
    atomicAdd(ap + 1, w * v.y);
    atomicAdd(ap + 2, w * v.z);
    atomicAdd(ap + 3, w * v.w);
}

// h = bf16(tanh(agg + bias)) on real columns, 0 on pad columns
__global__ __launch_bounds__(256) void k_biastanh(
    const float* __restrict__ agg, const float* __restrict__ bias,
    unsigned short* __restrict__ h, int total4, int Fpm1, int Fout)
{
    int gid = blockIdx.x * 256 + threadIdx.x;
    if (gid >= total4) return;
    float4 v = ((const float4*)agg)[gid];
    int f0 = (gid << 2) & Fpm1;
    ushort4 o;
    o.x = (f0 + 0 < Fout) ? f32_bf16(tanhf(v.x + bias[f0 + 0])) : (unsigned short)0;
    o.y = (f0 + 1 < Fout) ? f32_bf16(tanhf(v.y + bias[f0 + 1])) : (unsigned short)0;
    o.z = (f0 + 2 < Fout) ? f32_bf16(tanhf(v.z + bias[f0 + 2])) : (unsigned short)0;
    o.w = (f0 + 3 < Fout) ? f32_bf16(tanhf(v.w + bias[f0 + 3])) : (unsigned short)0;
    ((ushort4*)h)[gid] = o;
}

// ------------------------ edge concat + classifier -------------------------
// e_out[e] = concat(h[src], h[dst]) (24 f32), out[e] = e_out[e] @ Wc + bc (6)
__global__ __launch_bounds__(256) void k_edge_out(
    const unsigned short* __restrict__ h, const int* __restrict__ ei,
    const float* __restrict__ Wc, const float* __restrict__ bc,
    float* __restrict__ out, float* __restrict__ e_out, int E_)
{
    __shared__ float sw[144];
    __shared__ float sb[6];
    int tid = threadIdx.x;
    if (tid < 144) sw[tid] = Wc[tid];
    if (tid < 6)   sb[tid] = bc[tid];
    __syncthreads();

    int e = blockIdx.x * 256 + tid;
    if (e >= E_) return;
    int s = ei[e];
    int d = ei[E_ + e];
    const unsigned short* hs = h + ((size_t)s << 4);   // stride 16, 12 valid
    const unsigned short* hd = h + ((size_t)d << 4);
    uint4 q0 = *(const uint4*)hs;
    uint2 q1 = *(const uint2*)(hs + 8);
    uint4 p0 = *(const uint4*)hd;
    uint2 p1 = *(const uint2*)(hd + 8);

    float f[24] = {
        bf_lo(q0.x), bf_hi(q0.x), bf_lo(q0.y), bf_hi(q0.y),
        bf_lo(q0.z), bf_hi(q0.z), bf_lo(q0.w), bf_hi(q0.w),
        bf_lo(q1.x), bf_hi(q1.x), bf_lo(q1.y), bf_hi(q1.y),
        bf_lo(p0.x), bf_hi(p0.x), bf_lo(p0.y), bf_hi(p0.y),
        bf_lo(p0.z), bf_hi(p0.z), bf_lo(p0.w), bf_hi(p0.w),
        bf_lo(p1.x), bf_hi(p1.x), bf_lo(p1.y), bf_hi(p1.y) };

    float4* er = (float4*)(e_out + (size_t)e * 24);    // 96B rows, 16B aligned
    #pragma unroll
    for (int g = 0; g < 6; ++g) {
        float4 v = { f[4*g], f[4*g+1], f[4*g+2], f[4*g+3] };
        er[g] = v;
    }

    float acc[6];
    #pragma unroll
    for (int c = 0; c < 6; ++c) acc[c] = sb[c];
    #pragma unroll
    for (int j = 0; j < 24; ++j) {
        float fj = f[j];
        #pragma unroll
        for (int c = 0; c < 6; ++c) acc[c] += fj * sw[j * 6 + c];
    }
    float* orow = out + (size_t)e * 6;
    #pragma unroll
    for (int c = 0; c < 6; ++c) orow[c] = acc[c];
}

// ---------------------------------------------------------------------------
extern "C" void kernel_launch(void* const* d_in, const int* in_sizes, int n_in,
                              void* d_out, int out_size, void* d_ws, size_t ws_size,
                              hipStream_t stream) {
    const float* x   = (const float*)d_in[0];
    const int*   ei  = (const int*)  d_in[1];
    // d_in[2] = batch (unused)
    const float* W1  = (const float*)d_in[3];  const float* b1  = (const float*)d_in[4];
    const float* W2  = (const float*)d_in[5];  const float* b2  = (const float*)d_in[6];
    const float* W3  = (const float*)d_in[7];  const float* b3  = (const float*)d_in[8];
    const float* W4  = (const float*)d_in[9];  const float* b4  = (const float*)d_in[10];
    const float* W5  = (const float*)d_in[11]; const float* b5  = (const float*)d_in[12];
    const float* Wl1 = (const float*)d_in[13]; const float* bl1 = (const float*)d_in[14];
    const float* Wl2 = (const float*)d_in[15]; const float* bl2 = (const float*)d_in[16];
    const float* Wl3 = (const float*)d_in[17]; const float* bl3 = (const float*)d_in[18];
    const float* Wc  = (const float*)d_in[19]; const float* bc  = (const float*)d_in[20];

    const int N_ = in_sizes[0] / 128;
    const int E_ = in_sizes[1] / 2;

    float* ws   = (float*)d_ws;
    float* deg  = ws;
    float* dinv = ws + N_;
    float* xw   = dinv + N_;                       // N*256 f32 (GEMM out / scatter src)
    float* agg  = xw + (size_t)N_ * 256;           // N*256 f32 (aggregation accumulator)
    unsigned short* hA = (unsigned short*)(agg + (size_t)N_ * 256);  // N*256 bf16
    unsigned short* hB = hA + (size_t)N_ * 256;                      // N*64  bf16

    const dim3 B(256);
    auto cdiv = [](int a, int b) { return (a + b - 1) / b; };

    // --- symmetric GCN normalization ---
    k_fill1 <<<cdiv(N_, 256), B, 0, stream>>>(deg, N_);
    k_degacc<<<cdiv(E_, 256), B, 0, stream>>>(ei + E_, deg, E_);
    k_dinv  <<<cdiv(N_, 256), B, 0, stream>>>(deg, dinv, N_);

    // x -> bf16 (stored at hA; consumed by conv1's GEMM before hA is rewritten)
    k_cvt_bf16<<<cdiv(N_ * 32, 256), B, 0, stream>>>(x, hA, N_ * 32);

    const int gB = cdiv(N_, 128);   // WMMA GEMM grid

    auto gcn_agg = [&](int Fp, int lg4, const float* bias, int Fout,
                       unsigned short* h) {
        int Fp4 = Fp >> 2;
        int nt = N_ * Fp4;
        k_selfinit<<<cdiv(nt, 256), B, 0, stream>>>(xw, dinv, agg, nt, lg4);
        int et = E_ * Fp4;
        k_scatter <<<cdiv(et, 256), B, 0, stream>>>(xw, ei, dinv, agg, E_, Fp4, lg4, et);
        k_biastanh<<<cdiv(nt, 256), B, 0, stream>>>(agg, bias, h, nt, Fp - 1, Fout);
    };

    // conv1: x[N,128] @ W1[128,64]
    k_gemm_wmma<<<gB, B, 0, stream>>>(hA, 128, N_, W1, 128, 128, 64, 64, xw, nullptr, 64, nullptr, 0);
    gcn_agg(64, 4, b1, 64, hA);
    // conv2: h1 @ W2[64,64]
    k_gemm_wmma<<<gB, B, 0, stream>>>(hA, 64, N_, W2, 64, 64, 64, 64, xw, nullptr, 64, nullptr, 0);
    gcn_agg(64, 4, b2, 64, hB);
    // conv3: h2 @ W3[64,32]
    k_gemm_wmma<<<gB, B, 0, stream>>>(hB, 64, N_, W3, 64, 64, 32, 32, xw, nullptr, 32, nullptr, 0);
    gcn_agg(32, 3, b3, 32, hA);
    // conv4: h3 @ W4[32,32]
    k_gemm_wmma<<<gB, B, 0, stream>>>(hA, 32, N_, W4, 32, 32, 32, 32, xw, nullptr, 32, nullptr, 0);
    gcn_agg(32, 3, b4, 32, hB);
    // conv5: h4 @ W5[32,256]
    k_gemm_wmma<<<gB, B, 0, stream>>>(hB, 32, N_, W5, 32, 32, 256, 256, xw, nullptr, 256, nullptr, 0);
    gcn_agg(256, 6, b5, 256, hA);

    // lin1: h5[N,256] @ Wl1[256,24] + tanh  (pad 24->32), bf16 out
    k_gemm_wmma<<<gB, B, 0, stream>>>(hA, 256, N_, Wl1, 256, 256, 24, 32, nullptr, hB, 32, bl1, 1);
    // lin2: [N,24(p32)] @ Wl2[24,18] + tanh  (pad 18->32)
    k_gemm_wmma<<<gB, B, 0, stream>>>(hB, 32, N_, Wl2, 24, 32, 18, 32, nullptr, hA, 32, bl2, 1);
    // lin3: [N,18(p32)] @ Wl3[18,12] + tanh  (pad 12->16)
    k_gemm_wmma<<<gB, B, 0, stream>>>(hA, 32, N_, Wl3, 18, 32, 12, 16, nullptr, hB, 16, bl3, 1);

    // edge concat + classifier; d_out = [out (E*6) | e (E*24)]
    float* out0  = (float*)d_out;
    float* e_out = out0 + (size_t)E_ * 6;
    k_edge_out<<<cdiv(E_, 256), B, 0, stream>>>(hB, ei, Wc, bc, out0, e_out, E_);
}
